// Memory_11003706212844
// MI455X (gfx1250) — matrix operand order repaired and checked by reading, not compiled, not run on hardware
//
#include <hip/hip_runtime.h>

typedef __bf16 bf16_t;
typedef __bf16 bf16x16 __attribute__((ext_vector_type(16)));
typedef float floatx8 __attribute__((ext_vector_type(8)));

// ---------- WMMA fragment layout helpers (CDNA5 ISA 7.12.2) ----------
// A (16x32 bf16): lane L: row M = L%16, half = L/16; element i -> K = half*16 + i:
__device__ __forceinline__ int a_kmap(int half, int i){
  return half * 16 + i;
}
// B (32x16 bf16): lane L: col N = L%16, half = L/16; element i -> K = half*16 + i.
// Pre-swizzled B buffer: fragment (kt,nt) stored as 32 lanes x 16 contiguous bf16.
__device__ __forceinline__ size_t b_sw_idx(int k, int n, int NT){
  int kt = k >> 5, kk = k & 31, half = kk >> 4, i = kk & 15;
  int lane = half * 16 + (n & 15), nt = n >> 4;
  return ((size_t)((kt * NT + nt) * 32 + lane)) * 16 + i;
}

// ---------- wave reductions ----------
__device__ __forceinline__ float red_add16(float v){
  v += __shfl_xor(v, 1, 32); v += __shfl_xor(v, 2, 32);
  v += __shfl_xor(v, 4, 32); v += __shfl_xor(v, 8, 32);
  return v;
}
__device__ __forceinline__ float red_max16(float v){
  v = fmaxf(v, __shfl_xor(v, 1, 32)); v = fmaxf(v, __shfl_xor(v, 2, 32));
  v = fmaxf(v, __shfl_xor(v, 4, 32)); v = fmaxf(v, __shfl_xor(v, 8, 32));
  return v;
}
__device__ __forceinline__ float red_add32(float v){
  v += __shfl_xor(v, 1, 32); v += __shfl_xor(v, 2, 32);
  v += __shfl_xor(v, 4, 32); v += __shfl_xor(v, 8, 32);
  v += __shfl_xor(v, 16, 32);
  return v;
}

// ---------- LDS-free GEMM core: per-wave 16-row tile, NW n-tiles ----------
template<int KT, int NW, bool AF32>
__device__ __forceinline__ void gemm_frag(const void* Aptr, int lda, int colbase,
                                          int rowbase, const bf16_t* Bsw, int NT,
                                          int nt0, floatx8* acc){
  const int lane = threadIdx.x & 31;
  const int am = lane & 15, ah = lane >> 4;
  for (int kt = 0; kt < KT; ++kt){
    bf16x16 a;
    const int kb = colbase + kt * 32;
    if (AF32){
      const float* ar = (const float*)Aptr + (size_t)(rowbase + am) * lda + kb;
      #pragma unroll
      for (int i = 0; i < 16; ++i) a[i] = (bf16_t)ar[a_kmap(ah, i)];
    } else {
      const bf16_t* ar = (const bf16_t*)Aptr + (size_t)(rowbase + am) * lda + kb;
      #pragma unroll
      for (int i = 0; i < 16; ++i) a[i] = ar[a_kmap(ah, i)];
    }
    #pragma unroll
    for (int j = 0; j < NW; ++j){
      bf16x16 b = *(const bf16x16*)(Bsw + ((size_t)((kt * NT + nt0 + j) * 32 + lane)) * 16);
      acc[j] = __builtin_amdgcn_wmma_f32_16x16x32_bf16(false, a, false, b,
                                                       (short)0, acc[j], false, false);
    }
  }
}

#define ZERO_ACC(acc, NW)                    \
  _Pragma("unroll")                          \
  for (int j_ = 0; j_ < NW; ++j_)            \
    _Pragma("unroll")                        \
    for (int e_ = 0; e_ < 8; ++e_) acc[j_][e_] = 0.f;

// ---------- prep kernels (run every launch; weights are tiny) ----------
__global__ void k_zero(float* out){
  if (threadIdx.x < 2) out[16777216 + threadIdx.x] = 0.f;
}

__global__ __launch_bounds__(256) void p_swizzle(const float* src, bf16_t* dst,
    int K, int N, int scols, int transpose, int klim, int nlim){
  int idx = blockIdx.x * 256 + threadIdx.x;
  if (idx >= K * N) return;
  int k = idx / N, n = idx % N;
  float val = 0.f;
  if (k < klim && n < nlim)
    val = transpose ? src[(size_t)n * scols + k] : src[(size_t)k * scols + n];
  dst[b_sw_idx(k, n, N >> 4)] = (bf16_t)val;
}

__global__ void p_vnormplain(const float* value_param, float* value_norm){
  int r = threadIdx.x;
  if (r >= 112) return;
  const float* row = value_param + (size_t)r * 512;
  float s = 0.f;
  for (int d = 0; d < 512; ++d) s += row[d] * row[d];
  float inv = 1.f / fmaxf(sqrtf(s), 1e-12f);
  float* o = value_norm + (size_t)r * 512;
  for (int d = 0; d < 512; ++d) o[d] = row[d] * inv;
}

__global__ __launch_bounds__(256) void p_key(const float* key_param, bf16_t* keyB){
  int idx = blockIdx.x * 256 + threadIdx.x;    // 8 * 64 * 128
  int h = idx >> 13, rem = idx & 8191;
  int k = rem >> 7, n = rem & 127;
  float val = 0.f;
  if (n < 112){
    const float* row = key_param + ((size_t)h * 112 + n) * 64;
    float s = 0.f;
    #pragma unroll 8
    for (int d = 0; d < 64; ++d) s += row[d] * row[d];
    val = row[k] / fmaxf(sqrtf(s), 1e-12f);
  }
  keyB[(size_t)h * 8192 + b_sw_idx(k, n, 8)] = (bf16_t)val;
}

// Wcomb[h*128+j][o] = sum_d value_param[j][d] * out_w[o][h*512+d]  (padded j>=112 -> 0)
__global__ __launch_bounds__(256) void p_wcomb(const float* value_param,
                                               const float* out_w, bf16_t* wcombB){
  int idx = blockIdx.x * 256 + threadIdx.x;    // 1024 * 512
  int k = idx >> 9, n = idx & 511;
  int h = k >> 7, j = k & 127;
  float val = 0.f;
  if (j < 112){
    const float4* a = (const float4*)(value_param + (size_t)j * 512);
    const float4* w = (const float4*)(out_w + (size_t)n * 4096 + h * 512);
    float s = 0.f;
    for (int t = 0; t < 128; ++t){
      float4 av = a[t], wv = w[t];
      s += av.x * wv.x + av.y * wv.y + av.z * wv.z + av.w * wv.w;
    }
    val = s;
  }
  wcombB[b_sw_idx(k, n, 32)] = (bf16_t)val;
}

// ---------- main GEMM kernels: 128 token-rows per block, 8 waves ----------
__global__ __launch_bounds__(256) void k_qproj(const float* q, const bf16_t* qwT,
                                               const float* qb, bf16_t* qn){
  const int wave = threadIdx.x >> 5, lane = threadIdx.x & 31;
  const int rowbase = blockIdx.x * 128 + wave * 16;
  const int nl = lane & 15, half = lane >> 4;
  for (int h = 0; h < 8; ++h){                 // one head (64 cols) per chunk
    floatx8 acc[4]; ZERO_ACC(acc, 4)
    gemm_frag<16, 4, true>(q, 512, 0, rowbase, qwT, 32, h * 4, acc);
    float bj[4];
    #pragma unroll
    for (int j = 0; j < 4; ++j) bj[j] = qb[h * 64 + j * 16 + nl];
    #pragma unroll
    for (int e = 0; e < 8; ++e){
      float xv[4];
      #pragma unroll
      for (int j = 0; j < 4; ++j) xv[j] = acc[j][e] + bj[j];
      float s = xv[0]*xv[0] + xv[1]*xv[1] + xv[2]*xv[2] + xv[3]*xv[3];
      s = red_add16(s);                        // full 64-dim head norm
      float inv = 1.f / fmaxf(sqrtf(s), 1e-12f);
      int r = rowbase + e + 8 * half;
      bf16_t* dst = qn + (size_t)r * 512 + h * 64 + nl;
      #pragma unroll
      for (int j = 0; j < 4; ++j) dst[j * 16] = (bf16_t)(xv[j] * inv);
    }
  }
}

// softmax over 112 valid slots (tiles j=0..6), tile j=7 is padding -> write 0
__device__ __forceinline__ void softmax_store(floatx8* acc, int rowbase,
                                              bf16_t* dstbase, int ldd, int colbase){
  const int lane = threadIdx.x & 31;
  const int nl = lane & 15, half = lane >> 4;
  #pragma unroll
  for (int e = 0; e < 8; ++e){
    float mx = -3.4e38f;
    #pragma unroll
    for (int j = 0; j < 7; ++j) mx = fmaxf(mx, acc[j][e]);
    mx = red_max16(mx);
    float t[7]; float s = 0.f;
    #pragma unroll
    for (int j = 0; j < 7; ++j){
      t[j] = exp2f((acc[j][e] - mx) * 23.083120654223414f);  // RADIUS*log2(e)
      s += t[j];
    }
    s = red_add16(s);
    float inv = 1.f / s;
    int r = rowbase + e + 8 * half;
    bf16_t* dst = dstbase + (size_t)r * ldd + colbase + nl;
    #pragma unroll
    for (int j = 0; j < 7; ++j) dst[j * 16] = (bf16_t)(t[j] * inv);
    dst[112] = (bf16_t)0.f;
  }
}

__global__ __launch_bounds__(256) void k_simp(const bf16_t* qn, const bf16_t* keyB,
                                              bf16_t* addr){
  const int wave = threadIdx.x >> 5;
  const int h = blockIdx.y;
  const int rowbase = blockIdx.x * 128 + wave * 16;
  floatx8 acc[8]; ZERO_ACC(acc, 8)
  gemm_frag<2, 8, false>(qn, 512, h * 64, rowbase, keyB + (size_t)h * 8192, 8, 0, acc);
  softmax_store(acc, rowbase, addr, 1024, h * 128);
}

__global__ __launch_bounds__(256) void k_att(const bf16_t* addr, const bf16_t* wcB,
                                             const float* outb, float* fpred){
  const int wave = threadIdx.x >> 5, lane = threadIdx.x & 31;
  const int rowbase = blockIdx.x * 128 + wave * 16;
  const int nl = lane & 15, half = lane >> 4;
  for (int c = 0; c < 8; ++c){
    floatx8 acc[4]; ZERO_ACC(acc, 4)
    gemm_frag<32, 4, false>(addr, 1024, 0, rowbase, wcB, 32, c * 4, acc);
    float bj[4];
    #pragma unroll
    for (int j = 0; j < 4; ++j) bj[j] = outb[c * 64 + j * 16 + nl];
    #pragma unroll
    for (int e = 0; e < 8; ++e){
      int r = rowbase + e + 8 * half;
      float* dst = fpred + (size_t)r * 512 + c * 64 + nl;
      #pragma unroll
      for (int j = 0; j < 4; ++j) dst[j * 16] = acc[j][e] + bj[j];
    }
  }
}

__global__ __launch_bounds__(256) void k_vproj(const float* v, const bf16_t* vwT,
                                               const float* vb, float* vp){
  const int wave = threadIdx.x >> 5, lane = threadIdx.x & 31;
  const int rowbase = blockIdx.x * 128 + wave * 16;
  const int nl = lane & 15, half = lane >> 4;
  for (int c = 0; c < 8; ++c){
    floatx8 acc[4]; ZERO_ACC(acc, 4)
    gemm_frag<16, 4, true>(v, 512, 0, rowbase, vwT, 32, c * 4, acc);
    float bj[4];
    #pragma unroll
    for (int j = 0; j < 4; ++j) bj[j] = vb[c * 64 + j * 16 + nl];
    #pragma unroll
    for (int e = 0; e < 8; ++e){
      int r = rowbase + e + 8 * half;
      float* dst = vp + (size_t)r * 512 + c * 64 + nl;
      #pragma unroll
      for (int j = 0; j < 4; ++j) dst[j * 16] = acc[j][e] + bj[j];
    }
  }
}

__global__ __launch_bounds__(256) void k_simr(const bf16_t* vpn, const bf16_t* vnB,
                                              bf16_t* addrv){
  const int wave = threadIdx.x >> 5;
  const int rowbase = blockIdx.x * 128 + wave * 16;
  floatx8 acc[8]; ZERO_ACC(acc, 8)
  gemm_frag<16, 8, false>(vpn, 512, 0, rowbase, vnB, 8, 0, acc);
  softmax_store(acc, rowbase, addrv, 128, 0);
}

__global__ __launch_bounds__(256) void k_recon(const bf16_t* addrv, const bf16_t* valB,
                                               float* ftgt){
  const int wave = threadIdx.x >> 5, lane = threadIdx.x & 31;
  const int rowbase = blockIdx.x * 128 + wave * 16;
  const int nl = lane & 15, half = lane >> 4;
  for (int c = 0; c < 8; ++c){
    floatx8 acc[4]; ZERO_ACC(acc, 4)
    gemm_frag<4, 4, false>(addrv, 128, 0, rowbase, valB, 32, c * 4, acc);
    #pragma unroll
    for (int e = 0; e < 8; ++e){
      int r = rowbase + e + 8 * half;
      float* dst = ftgt + (size_t)r * 512 + c * 64 + nl;
      #pragma unroll
      for (int j = 0; j < 4; ++j) dst[j * 16] = acc[j][e];
    }
  }
}

// ---------- row-wise kernels: one wave per 512-wide row ----------
__global__ __launch_bounds__(256) void k_ln(float* data, const float* g, const float* b){
  const int wave = threadIdx.x >> 5, lane = threadIdx.x & 31;
  size_t row = (size_t)blockIdx.x * 8 + wave;
  float* p = data + row * 512;
  float x[16]; float s = 0.f, sq = 0.f;
  #pragma unroll
  for (int i = 0; i < 16; ++i){ x[i] = p[lane + 32 * i]; s += x[i]; sq += x[i] * x[i]; }
  s = red_add32(s); sq = red_add32(sq);
  float mu = s * (1.f / 512.f);
  float var = sq * (1.f / 512.f) - mu * mu;
  float rs = rsqrtf(var + 1e-5f);
  #pragma unroll
  for (int i = 0; i < 16; ++i){ int c = lane + 32 * i; p[c] = (x[i] - mu) * rs * g[c] + b[c]; }
}

__global__ __launch_bounds__(256) void k_vnormrow(const float* vp, bf16_t* vpn){
  const int wave = threadIdx.x >> 5, lane = threadIdx.x & 31;
  size_t row = (size_t)blockIdx.x * 8 + wave;
  const float* p = vp + row * 512;
  bf16_t* q = vpn + row * 512;
  float x[16]; float s = 0.f;
  #pragma unroll
  for (int i = 0; i < 16; ++i){ x[i] = p[lane + 32 * i]; s += x[i] * x[i]; }
  s = red_add32(s);
  float inv = 1.f / fmaxf(sqrtf(s), 1e-12f);
  #pragma unroll
  for (int i = 0; i < 16; ++i) q[lane + 32 * i] = (bf16_t)(x[i] * inv);
}

__global__ __launch_bounds__(256) void k_reconfinish(float* ar, const float* vin,
    const float* g, const float* b, float* loss){
  const int wave = threadIdx.x >> 5, lane = threadIdx.x & 31;
  size_t row = (size_t)blockIdx.x * 8 + wave;
  float* pa = ar + row * 512;
  const float* pv = vin + row * 512;
  float a[16]; float sa = 0.f, na = 0.f, nv = 0.f, num = 0.f;
  #pragma unroll
  for (int i = 0; i < 16; ++i){
    a[i] = pa[lane + 32 * i];
    float vv = pv[lane + 32 * i];
    sa += a[i]; na += a[i] * a[i]; nv += vv * vv; num += a[i] * vv;
  }
  sa = red_add32(sa); na = red_add32(na); nv = red_add32(nv); num = red_add32(num);
  if (lane == 0){
    float den = fmaxf(sqrtf(na) * sqrtf(nv), 1e-8f);
    atomicAdd(loss, fabsf(1.f - num / den) * (1.f / 16384.f));
  }
  float mu = sa * (1.f / 512.f);
  float var = na * (1.f / 512.f) - mu * mu;
  float rs = rsqrtf(var + 1e-5f);
  #pragma unroll
  for (int i = 0; i < 16; ++i){ int c = lane + 32 * i; pa[c] = (a[i] - mu) * rs * g[c] + b[c]; }
}

__global__ __launch_bounds__(256) void k_contrastive(const float* vn, float* loss){
  int idx = blockIdx.x * 256 + threadIdx.x;
  float c = 0.f;
  if (idx < 112 * 112){
    int i = idx / 112, jj = idx % 112;
    const float4* a = (const float4*)(vn + (size_t)i * 512);
    const float4* b = (const float4*)(vn + (size_t)jj * 512);
    float s = 0.f;
    for (int t = 0; t < 128; ++t){
      float4 av = a[t], bv = b[t];
      s += av.x * bv.x + av.y * bv.y + av.z * bv.z + av.w * bv.w;
    }
    c = fabsf(((i == jj) ? 1.f : 0.f) - s);
  }
  c = red_add32(c);
  if ((threadIdx.x & 31) == 0) atomicAdd(loss, c * 0.01f);
}

// ---------- host ----------
extern "C" void kernel_launch(void* const* d_in, const int* in_sizes, int n_in,
                              void* d_out, int out_size, void* d_ws, size_t ws_size,
                              hipStream_t stream){
  (void)in_sizes; (void)n_in; (void)out_size; (void)ws_size;
  const float* q     = (const float*)d_in[0];
  const float* v     = (const float*)d_in[1];
  const float* keyp  = (const float*)d_in[2];
  const float* valp  = (const float*)d_in[3];
  const float* qw    = (const float*)d_in[4];
  const float* qb    = (const float*)d_in[5];
  const float* vw    = (const float*)d_in[6];
  const float* vb    = (const float*)d_in[7];
  const float* outw  = (const float*)d_in[8];
  const float* outb  = (const float*)d_in[9];
  const float* n2g   = (const float*)d_in[10];
  const float* n2b   = (const float*)d_in[11];
  const float* n3g   = (const float*)d_in[12];
  const float* n3b   = (const float*)d_in[13];

  char* W = (char*)d_ws;
  bf16_t* qwT    = (bf16_t*)(W + 0);          // 512x512  bf16  (512 KB)
  bf16_t* vwT    = (bf16_t*)(W + 524288);     // 512x512  bf16
  bf16_t* keyB   = (bf16_t*)(W + 1048576);    // 8 x (64x128) bf16
  bf16_t* vnB    = (bf16_t*)(W + 1179648);    // 512x128  bf16
  bf16_t* valB   = (bf16_t*)(W + 1310720);    // 128x512  bf16
  bf16_t* wcB    = (bf16_t*)(W + 1441792);    // 1024x512 bf16 (1 MB)
  float*  vnorm  = (float*) (W + 2490368);    // 112x512  f32
  bf16_t* qn     = (bf16_t*)(W + 2719744);    // 16384x512 bf16 (16.8 MB) -> later vpn
  bf16_t* addr   = (bf16_t*)(W + 19496960);   // 16384x1024 bf16 (33.5 MB)
  float*  vp     = (float*) (W + 19496960);   // reuse after k_att consumes addr
  bf16_t* addrv  = (bf16_t*)(W + 53051392);   // 16384x128 bf16 (4.2 MB)

  float* out   = (float*)d_out;
  float* fpred = out;                 // [16384,512] raw att, LN in place
  float* ftgt  = out + 8388608;       // [16384,512] raw att_recon, LN in place

  k_zero<<<1, 64, 0, stream>>>(out);

  // prep / swizzle (cheap, every launch)
  p_swizzle<<<1024, 256, 0, stream>>>(qw, qwT, 512, 512, 512, 1, 512, 512);
  p_swizzle<<<1024, 256, 0, stream>>>(vw, vwT, 512, 512, 512, 1, 512, 512);
  p_vnormplain<<<1, 128, 0, stream>>>(valp, vnorm);
  p_swizzle<<<256, 256, 0, stream>>>(vnorm, vnB, 512, 128, 512, 1, 512, 112);
  p_swizzle<<<256, 256, 0, stream>>>(valp, valB, 128, 512, 512, 0, 112, 512);
  p_key<<<256, 256, 0, stream>>>(keyp, keyB);
  p_wcomb<<<2048, 256, 0, stream>>>(valp, outw, wcB);

  // predict path
  k_qproj<<<128, 256, 0, stream>>>(q, qwT, qb, qn);
  k_simp<<<dim3(128, 8), 256, 0, stream>>>(qn, keyB, addr);
  k_att<<<128, 256, 0, stream>>>(addr, wcB, outb, fpred);
  k_ln<<<2048, 256, 0, stream>>>(fpred, n2g, n2b);

  // recon path (reuses addr region for vp, qn region for vpn)
  k_vproj<<<128, 256, 0, stream>>>(v, vwT, vb, vp);
  k_vnormrow<<<2048, 256, 0, stream>>>(vp, qn);
  k_simr<<<128, 256, 0, stream>>>(qn, vnB, addrv);
  k_recon<<<128, 256, 0, stream>>>(addrv, valB, ftgt);
  k_reconfinish<<<2048, 256, 0, stream>>>(ftgt, v, n3g, n3b, out + 16777216);
  k_contrastive<<<49, 256, 0, stream>>>(vnorm, out + 16777217);
}
